// GNNPredictor_30975304139273
// MI455X (gfx1250) — compile-verified, matrix operand order checked by reading, not run on hardware
//
#include <hip/hip_runtime.h>
#include <hip/hip_bf16.h>

#define N_AGENTS 50
#define TSTEPS   110
#define FSTATE   6
#define MPAD     64      // 50 agents padded to 4 row-tiles of 16
#define DMODEL   128

typedef __attribute__((ext_vector_type(16))) _Float16 f16x16;
typedef __attribute__((ext_vector_type(8)))  _Float16 f16x8;
typedef __attribute__((ext_vector_type(8)))  float    f32x8;

// Load one lane's slice of a 16x32 (f16) WMMA operand from LDS.
// CDNA5 16-bit A/B layout: lane L owns row/col (L&15), half = L>>4,
// element e -> K = (e/8)*16 + half*8 + (e%8). The two 8-element K-runs
// are contiguous in memory -> two ds_load_b128.
__device__ __forceinline__ f16x16 load_frag_k32(const _Float16* __restrict__ base,
                                                int ld, int row, int k0, int half) {
  const _Float16* p = base + row * ld + k0 + half * 8;
  f16x8 lo = *(const f16x8*)(p);        // K = half*8 + 0..7
  f16x8 hi = *(const f16x8*)(p + 16);   // K = 16 + half*8 + 0..7
  f16x16 r;
#pragma unroll
  for (int i = 0; i < 8; ++i) { r[i] = lo[i]; r[i + 8] = hi[i]; }
  return r;
}

// C(64 x 16*N_TILES) = A(64 x KDIM) * B(KDIM x N), B given transposed (BT[n][k]).
// 8 waves; wave w owns row-tile (w&3) -> its A fragments are loaded once and
// reused across its N_TILES/2 column tiles. Fully unrolled (compile-time shapes),
// branch-free epilogue via template flags.
template<int N_TILES, int KDIM, bool HAS_BIAS, bool ROW_OUT, bool COL_OUT>
__device__ __forceinline__ void gemm_wmma_64xN(
    const _Float16* __restrict__ A, int lda,
    const _Float16* __restrict__ BT, int ldb,
    const float* __restrict__ bias,
    _Float16* __restrict__ Crow, int ldcr,
    _Float16* __restrict__ CcolT, int ldcc)
{
  const int wave = threadIdx.x >> 5;
  const int lane = threadIdx.x & 31;
  const int half = lane >> 4;
  const int mn   = lane & 15;
  const int tm   = wave & 3;                 // fixed row tile per wave

  f16x16 afr[KDIM / 32];                      // A fragments, reused per wave
#pragma unroll
  for (int kk = 0; kk < KDIM / 32; ++kk)
    afr[kk] = load_frag_k32(A, lda, tm * 16 + mn, kk * 32, half);

#pragma unroll
  for (int i = 0; i < N_TILES / 2; ++i) {
    const int tn = (wave >> 2) + 2 * i;
    f32x8 acc = {};
#pragma unroll
    for (int kk = 0; kk < KDIM / 32; ++kk) {
      f16x16 bf = load_frag_k32(BT, ldb, tn * 16 + mn, kk * 32, half);
      acc = __builtin_amdgcn_wmma_f32_16x16x32_f16(false, afr[kk], false, bf,
                                                   (short)0, acc, false, false);
    }
    const int n  = tn * 16 + mn;
    const float bv = HAS_BIAS ? bias[n] : 0.0f;
    if (COL_OUT) {
      // m = tm*16 + 8*half + v is contiguous over v -> single b128 store
      f16x8 pack;
#pragma unroll
      for (int v = 0; v < 8; ++v) pack[v] = (_Float16)(acc[v] + bv);
      *(f16x8*)(CcolT + n * ldcc + tm * 16 + 8 * half) = pack;
    }
    if (ROW_OUT) {
#pragma unroll
      for (int v = 0; v < 8; ++v)
        Crow[(tm * 16 + v + 8 * half) * ldcr + n] = (_Float16)(acc[v] + bv);
    }
  }
}

__global__ __launch_bounds__(256) void gnn_predictor_kernel(
    const float* __restrict__ x,
    const float* __restrict__ W1, const float* __restrict__ b1,
    const float* __restrict__ W2, const float* __restrict__ b2,
    const float* __restrict__ Wg,
    const float* __restrict__ a1, const float* __restrict__ a2,
    const float* __restrict__ Wo1, const float* __restrict__ bo1,
    const float* __restrict__ Wo2, const float* __restrict__ bo2,
    float* __restrict__ out)
{
  __shared__ __align__(16) _Float16 s_nfA[MPAD * DMODEL];   // 16 KB
  __shared__ __align__(16) _Float16 s_nfB[MPAD * DMODEL];   // 16 KB
  __shared__ __align__(16) _Float16 s_hT[DMODEL * MPAD];    // 16 KB (h transposed)
  __shared__ __align__(16) _Float16 s_wT[DMODEL * DMODEL];  // 32 KB (weight^T stage)
  __shared__ __align__(16) _Float16 s_attn[MPAD * MPAD];    //  8 KB
  __shared__ float s_cur[MPAD * FSTATE];
  __shared__ float s_s1[MPAD];
  __shared__ float s_s2[MPAD];
  __shared__ float s_hidden[512];

  const int b   = blockIdx.x;
  const int tid = threadIdx.x;

  // ---- load only the live slice of x: last timestep (50x6), zero-pad to 64 ----
  for (int idx = tid; idx < MPAD * FSTATE; idx += 256) {
    const int m = idx / FSTATE, f = idx % FSTATE;
    s_cur[idx] = (m < N_AGENTS)
        ? x[(size_t)b * (N_AGENTS * TSTEPS * FSTATE)
            + (size_t)m * (TSTEPS * FSTATE) + (TSTEPS - 1) * FSTATE + f]
        : 0.0f;
  }
  __syncthreads();

  // ---- nf0 = relu(cur @ W1 + b1): K=6 too small for WMMA -> VALU ----
  for (int idx = tid; idx < MPAD * DMODEL; idx += 256) {
    const int m = idx >> 7, j = idx & 127;
    float s = b1[j];
#pragma unroll
    for (int f = 0; f < FSTATE; ++f) s += s_cur[m * FSTATE + f] * W1[f * DMODEL + j];
    s_nfA[idx] = (_Float16)fmaxf(s, 0.0f);
  }
  __syncthreads();

  // ---- stage W2^T as f16 ----
  for (int idx = tid; idx < DMODEL * DMODEL; idx += 256) {
    const int k = idx >> 7, n = idx & 127;
    s_wT[n * DMODEL + k] = (_Float16)W2[idx];
  }
  __syncthreads();

  // ---- nf = nf0 @ W2 + b2 (WMMA) ----
  gemm_wmma_64xN<8, DMODEL, true, true, false>(s_nfA, DMODEL, s_wT, DMODEL, b2,
                                               s_nfB, DMODEL, nullptr, 0);
  __syncthreads();

  _Float16* nf_cur = s_nfB;
  _Float16* nf_alt = s_nfA;

  // ---- 3 GAT layers ----
  for (int l = 0; l < 3; ++l) {
    const float* Wl = Wg + (size_t)l * DMODEL * DMODEL;

    // prefetch next layer's weights toward the WGP while this layer computes
    if (l < 2) {
      const float* Wnext = Wl + DMODEL * DMODEL;
      for (int c = tid; c < 512; c += 256)           // 512 x 128B lines = 64 KB
        __builtin_prefetch(Wnext + c * 32, 0, 0);    // global_prefetch_b8
    }

    for (int idx = tid; idx < DMODEL * DMODEL; idx += 256) {
      const int k = idx >> 7, n = idx & 127;
      s_wT[n * DMODEL + k] = (_Float16)Wl[idx];
    }
    __syncthreads();

    // h^T = (nf @ Wg[l])^T : transposed epilogue feeds the attn GEMM B operand
    gemm_wmma_64xN<8, DMODEL, false, false, true>(nf_cur, DMODEL, s_wT, DMODEL,
                                                  nullptr, nullptr, 0, s_hT, MPAD);
    __syncthreads();

    // s1[i] = h[i,:]·a1 ; s2[i] = h[i,:]·a2  (one thread per node row)
    if (tid < MPAD) {
      float v1 = 0.f, v2 = 0.f;
      for (int d = 0; d < DMODEL; ++d) {
        const float hv = (float)s_hT[d * MPAD + tid];
        v1 += hv * a1[l * DMODEL + d];
        v2 += hv * a2[l * DMODEL + d];
      }
      s_s1[tid] = v1; s_s2[tid] = v2;
    }
    __syncthreads();

    // row softmax of leaky_relu(s1_i + s2_j, 0.2), diag = -1e9, padded cols = 0
    if (tid < MPAD) {
      const int i = tid;
      if (i < N_AGENTS) {
        const float si = s_s1[i];
        float mx = -3.0e38f;
        for (int j = 0; j < N_AGENTS; ++j) {
          const float v = si + s_s2[j];
          const float e = (j == i) ? -1.0e9f : (v > 0.f ? v : 0.2f * v);
          mx = fmaxf(mx, e);
        }
        float sum = 0.f;
        for (int j = 0; j < N_AGENTS; ++j) {
          const float v = si + s_s2[j];
          const float e = (j == i) ? -1.0e9f : (v > 0.f ? v : 0.2f * v);
          sum += __expf(e - mx);
        }
        const float inv = 1.0f / sum;
        for (int j = 0; j < MPAD; ++j) {
          float w = 0.f;
          if (j < N_AGENTS && j != i) {
            const float v = si + s_s2[j];
            const float e = v > 0.f ? v : 0.2f * v;
            w = __expf(e - mx) * inv;
          }
          s_attn[i * MPAD + j] = (_Float16)w;   // zero pad cols kill padded h rows
        }
      } else {
        for (int j = 0; j < MPAD; ++j) s_attn[i * MPAD + j] = (_Float16)0.f;
      }
    }
    __syncthreads();

    // nf_next = attn @ h   (A = attn 64x64, B^T = h^T, K = 64)
    gemm_wmma_64xN<8, MPAD, false, true, false>(s_attn, MPAD, s_hT, MPAD,
                                                nullptr, nf_alt, DMODEL, nullptr, 0);
    __syncthreads();

    _Float16* tmp = nf_cur; nf_cur = nf_alt; nf_alt = tmp;
  }

  // ---- ego (row 0) output MLP: 128 -> 512 -> 120, f32 VALU ----
  for (int j = tid; j < 512; j += 256) {
    float s = bo1[j];
    for (int d = 0; d < DMODEL; ++d) s += (float)nf_cur[d] * Wo1[d * 512 + j];
    s_hidden[j] = fmaxf(s, 0.0f);
  }
  __syncthreads();
  for (int k = tid; k < 120; k += 256) {
    float s = bo2[k];
    for (int d = 0; d < 512; ++d) s += s_hidden[d] * Wo2[d * 120 + k];
    out[(size_t)b * 120 + k] = s;
  }
}

extern "C" void kernel_launch(void* const* d_in, const int* in_sizes, int n_in,
                              void* d_out, int out_size, void* d_ws, size_t ws_size,
                              hipStream_t stream) {
  (void)n_in; (void)out_size; (void)d_ws; (void)ws_size;
  const float* x   = (const float*)d_in[0];
  const float* W1  = (const float*)d_in[1];
  const float* b1  = (const float*)d_in[2];
  const float* W2  = (const float*)d_in[3];
  const float* b2  = (const float*)d_in[4];
  // d_in[5..8] = We1, be1, We2, be2: dead in the reference (ef computed then deleted)
  const float* Wg  = (const float*)d_in[9];
  const float* a1  = (const float*)d_in[10];
  const float* a2  = (const float*)d_in[11];
  const float* Wo1 = (const float*)d_in[12];
  const float* bo1 = (const float*)d_in[13];
  const float* Wo2 = (const float*)d_in[14];
  const float* bo2 = (const float*)d_in[15];
  float* out = (float*)d_out;

  const int B = in_sizes[0] / (N_AGENTS * TSTEPS * FSTATE);   // 1024
  gnn_predictor_kernel<<<B, 256, 0, stream>>>(x, W1, b1, W2, b2, Wg, a1, a2,
                                              Wo1, bo1, Wo2, bo2, out);
}
